// EdgeWeightLearner_8976481648848
// MI455X (gfx1250) — matrix operand order repaired and checked by reading, not compiled, not run on hardware
//
#include <hip/hip_runtime.h>
#include <math.h>

typedef __attribute__((ext_vector_type(16))) _Float16 v16h;
typedef __attribute__((ext_vector_type(8)))  float    v8f;

#define IN_CH 64

// ---------------------------------------------------------------------------
// Kernel 1: per-node scores via WMMA.
//   s_row[n] = x[n] . w_lin[0:64]
//   s_col[n] = x[n] . w_lin[64:128]
// One wave32 per 16-node tile: D(16x16) = A(16x32) x B(32x16) + C, two K=32
// steps covering IN_CH=64. B holds w_row in column 0, w_col in column 1,
// zeros elsewhere.  Branch-free loads (clamped addresses + select) keep EXEC
// full; full tiles store results as two b128 stores per live lane.
// ---------------------------------------------------------------------------
__global__ __launch_bounds__(256) void node_scores_wmma(
    const float* __restrict__ x, const float* __restrict__ w,
    float* __restrict__ s_row, float* __restrict__ s_col, int n_nodes)
{
    const int lane = threadIdx.x & 31;
    const int wave = threadIdx.x >> 5;
    const int tile = blockIdx.x * 8 + wave;
    const int n_tiles = (n_nodes + 15) >> 4;
    if (tile >= n_tiles) return;          // uniform per wave -> EXEC stays full
    const int base = tile << 4;

    const int half = lane >> 4;           // lane group: 0 = lanes 0-15, 1 = 16-31
    const int m    = lane & 15;           // A row / B-C-D column index
    const bool live_col = (m < 2);        // B columns 0/1 carry w_row/w_col

    int node = base + m;
    if (node >= n_nodes) node = n_nodes - 1;   // clamp loads (stores guarded)
    const float* __restrict__ xr = x + (size_t)node * IN_CH;

    // Branch-free B base: lanes m>=2 alias column 0's range, result zeroed.
    const int wcol = live_col ? m * IN_CH : 0;

    v8f acc = {};
#pragma unroll
    for (int step = 0; step < 2; ++step) {
        const int koff = step * 32;

        // A 16x32 f16 layout (ISA 7.12.2): lane group h holds, for its row,
        // K = { h*8 .. h*8+7 , 16+h*8 .. 16+h*8+7 }: two contiguous float4
        // pairs, converted to f16.
        const float4* x4 = (const float4*)xr;
        const int c0 = (koff + half * 8) >> 2;
        const int c1 = (koff + 16 + half * 8) >> 2;
        float4 p0 = x4[c0], p1 = x4[c0 + 1];
        float4 p2 = x4[c1], p3 = x4[c1 + 1];

        v16h a;
        a[0]  = (_Float16)p0.x; a[1]  = (_Float16)p0.y;
        a[2]  = (_Float16)p0.z; a[3]  = (_Float16)p0.w;
        a[4]  = (_Float16)p1.x; a[5]  = (_Float16)p1.y;
        a[6]  = (_Float16)p1.z; a[7]  = (_Float16)p1.w;
        a[8]  = (_Float16)p2.x; a[9]  = (_Float16)p2.y;
        a[10] = (_Float16)p2.z; a[11] = (_Float16)p2.w;
        a[12] = (_Float16)p3.x; a[13] = (_Float16)p3.y;
        a[14] = (_Float16)p3.z; a[15] = (_Float16)p3.w;

        // B 32x16 f16: lane holds column n=m, K = { half*8 .. half*8+7,
        // 16+half*8 .. 16+half*8+7 } (same order as A).  Unconditional b128
        // loads; dead columns zeroed via select (no EXEC churn).
        const float4* w4 = (const float4*)(w + wcol + koff + half * 8);
        float4 q0 = w4[0], q1 = w4[1], q2 = w4[4], q3 = w4[5];
        const float zm = live_col ? 1.0f : 0.0f;

        v16h b;
        b[0]  = (_Float16)(q0.x * zm); b[1]  = (_Float16)(q0.y * zm);
        b[2]  = (_Float16)(q0.z * zm); b[3]  = (_Float16)(q0.w * zm);
        b[4]  = (_Float16)(q1.x * zm); b[5]  = (_Float16)(q1.y * zm);
        b[6]  = (_Float16)(q1.z * zm); b[7]  = (_Float16)(q1.w * zm);
        b[8]  = (_Float16)(q2.x * zm); b[9]  = (_Float16)(q2.y * zm);
        b[10] = (_Float16)(q2.z * zm); b[11] = (_Float16)(q2.w * zm);
        b[12] = (_Float16)(q3.x * zm); b[13] = (_Float16)(q3.y * zm);
        b[14] = (_Float16)(q3.z * zm); b[15] = (_Float16)(q3.w * zm);

        acc = __builtin_amdgcn_wmma_f32_16x16x32_f16(
            /*neg_a=*/false, a, /*neg_b=*/false, b,
            /*c_mod=*/(short)0, acc, /*reuse_a=*/false, /*reuse_b=*/false);
    }

    // D layout: lane = half*16 + N, VGPR r = (M = half*8 + r, N).
    // N=0 -> s_row, N=1 -> s_col: lanes 0,1,16,17 store 8 contiguous floats
    // each.  Full tiles (the only case at N=50000) use two b128 stores.
    if (m < 2) {
        float* dst = (m == 0 ? s_row : s_col) + base + half * 8;
        if (base + 16 <= n_nodes) {
            float4 lo, hi;
            lo.x = acc[0]; lo.y = acc[1]; lo.z = acc[2]; lo.w = acc[3];
            hi.x = acc[4]; hi.y = acc[5]; hi.z = acc[6]; hi.w = acc[7];
            ((float4*)dst)[0] = lo;
            ((float4*)dst)[1] = hi;
        } else {
            const int lim = n_nodes - (base + half * 8);
#pragma unroll
            for (int r = 0; r < 8; ++r)
                if (r < lim) dst[r] = acc[r];
        }
    }
}

// ---------------------------------------------------------------------------
// Kernel 2: per-edge sigmoid weight.  s tables (400 KB) are L2-resident, so
// the two gathers are cheap; streaming cost is edge_index (2x4 B/edge).
// ---------------------------------------------------------------------------
__global__ __launch_bounds__(256) void edge_weights_kernel(
    const int* __restrict__ ei, const float* __restrict__ s_row,
    const float* __restrict__ s_col, float* __restrict__ wts, int n_edges)
{
    int e = blockIdx.x * blockDim.x + threadIdx.x;
    if (e >= n_edges) return;
    int r = ei[e];                 // edge_index[0][e]
    int c = ei[n_edges + e];       // edge_index[1][e]
    float logit = s_row[r] + s_col[c];
    wts[e] = 1.0f / (1.0f + __expf(-logit));
}

// ---------------------------------------------------------------------------
// Kernel 3: out[e] = w[e] * w[fri[e]].  weights (3.2 MB) stays in the 192 MB
// L2 between kernels 2 and 3, so the gather hits L2, not HBM.
// ---------------------------------------------------------------------------
__global__ __launch_bounds__(256) void pair_mul_kernel(
    const float* __restrict__ wts, const int* __restrict__ fri,
    float* __restrict__ out, int n_edges)
{
    int e = blockIdx.x * blockDim.x + threadIdx.x;
    if (e >= n_edges) return;
    out[e] = wts[e] * wts[fri[e]];
}

extern "C" void kernel_launch(void* const* d_in, const int* in_sizes, int n_in,
                              void* d_out, int out_size, void* d_ws, size_t ws_size,
                              hipStream_t stream)
{
    const float* x   = (const float*)d_in[0];   // (N_NODES, 64) f32
    const float* w   = (const float*)d_in[1];   // (128,) f32
    const int*   ei  = (const int*)d_in[2];     // (2, N_EDGES) int
    const int*   fri = (const int*)d_in[3];     // (N_EDGES,) int
    float* out = (float*)d_out;

    const int n_nodes = in_sizes[0] / IN_CH;
    const int n_edges = in_sizes[3];

    // Workspace: s_row (n_nodes) | s_col (n_nodes) | weights (n_edges) f32
    float* s_row = (float*)d_ws;
    float* s_col = s_row + n_nodes;
    float* wts   = s_col + n_nodes;

    const int n_tiles = (n_nodes + 15) / 16;     // 16 nodes per wave
    const int blocks1 = (n_tiles + 7) / 8;       // 8 waves per 256-thread block
    node_scores_wmma<<<blocks1, 256, 0, stream>>>(x, w, s_row, s_col, n_nodes);

    const int blocks2 = (n_edges + 255) / 256;
    edge_weights_kernel<<<blocks2, 256, 0, stream>>>(ei, s_row, s_col, wts, n_edges);
    pair_mul_kernel<<<blocks2, 256, 0, stream>>>(wts, fri, out, n_edges);
}
